// MultiHeadAttention_75445395522226
// MI455X (gfx1250) — matrix operand order repaired and checked by reading, not compiled, not run on hardware
//
#include <hip/hip_runtime.h>

#define D_MODEL 768
#define NUM_HEADS 12
#define HEAD_DIM 64
#define SEQ 4096
#define BATCH 2
#define M_ROWS (BATCH * SEQ)   /* 8192 */
#define N_QKV (3 * D_MODEL)    /* 2304 */
#define LDS_LD 72              /* padded row stride (halves): 36 dwords, bank-conflict free */

typedef __attribute__((ext_vector_type(16))) _Float16 v16h;
typedef __attribute__((ext_vector_type(8)))  float    v8f;

// ---- WMMA operand loaders (contiguous k order: element s holds k = 16*(lane/16) + s) ----------

// A operand: 16x32 f16 tile, row-major source with leading dim ld.
// lane m = lane&15; VGPR v holds K pair: 16*(lane/16) + 2*v
template <typename PtrT>
__device__ __forceinline__ v16h wmma_load_a(PtrT p, int ld, int lane) {
  const int m  = lane & 15;
  const int hs = lane >> 4;
  v16h a;
#pragma unroll
  for (int v = 0; v < 8; ++v) {
    const int k = (hs << 4) + (v << 1);
    a[2 * v]     = p[(size_t)m * ld + k];
    a[2 * v + 1] = p[(size_t)m * ld + k + 1];
  }
  return a;
}

// B operand: 32x16 f16 tile, row-major K x N source with leading dim ld.
// lane n = lane&15; lanes 0-15 hold K=0..15, lanes 16-31 hold K=16..31.
template <typename PtrT>
__device__ __forceinline__ v16h wmma_load_b(PtrT p, int ld, int lane) {
  const int n  = lane & 15;
  const int kb = (lane >> 4) << 4;
  v16h b;
#pragma unroll
  for (int v = 0; v < 8; ++v) {
    const int k = kb + (v << 1);
    b[2 * v]     = p[(size_t)k * ld + n];
    b[2 * v + 1] = p[(size_t)(k + 1) * ld + n];
  }
  return b;
}

// B operand for transposed source: logical B[k][n] = p[n*ld + k] (per-lane
// contiguous pairs -> b128-coalescible).
template <typename PtrT>
__device__ __forceinline__ v16h wmma_load_bT(PtrT p, int ld, int lane) {
  const int n  = lane & 15;
  const int kb = (lane >> 4) << 4;
  v16h b;
#pragma unroll
  for (int v = 0; v < 8; ++v) {
    const int k = kb + (v << 1);
    b[2 * v]     = p[(size_t)n * ld + k];
    b[2 * v + 1] = p[(size_t)n * ld + k + 1];
  }
  return b;
}

__device__ __forceinline__ v8f wmma_f16(v16h a, v16h b, v8f c) {
  return __builtin_amdgcn_wmma_f32_16x16x32_f16(false, a, false, b, (short)0, c,
                                                false, false);
}

// ---- f32 -> f16 conversion ------------------------------------------------
__global__ void f32_to_f16_kernel(const float* __restrict__ src,
                                  _Float16* __restrict__ dst, int n) {
  int i = blockIdx.x * blockDim.x + threadIdx.x;
  const int stride = gridDim.x * blockDim.x;
  for (; i < n; i += stride) dst[i] = (_Float16)src[i];
}

// f32 [K][N] -> f16 [N][K] (weights: transpose once so GEMM B-operands are
// contiguous per lane)
__global__ void transpose_f32_to_f16_kernel(const float* __restrict__ src,
                                            _Float16* __restrict__ dst, int K,
                                            int N) {
  int i = blockIdx.x * blockDim.x + threadIdx.x;
  const int stride = gridDim.x * blockDim.x;
  const int total = K * N;
  for (; i < total; i += stride) {
    const int n = i / K, k = i - n * K;
    dst[i] = (_Float16)src[(size_t)k * N + n];
  }
}

// ---- QKV projection: qkv = Xh @ W + b, scattered to Q/K/V [B,H,T,64] ------
__global__ __launch_bounds__(128) void qkv_gemm_kernel(
    const _Float16* __restrict__ Xh, const _Float16* __restrict__ WaT,
    const float* __restrict__ bias, _Float16* __restrict__ Qh,
    _Float16* __restrict__ Kh, _Float16* __restrict__ Vh) {
  const int lane = threadIdx.x & 31;
  const int wave = threadIdx.x >> 5;
  const int m0 = blockIdx.y * 64 + (wave >> 1) * 32;
  const int n0 = blockIdx.x * 64 + (wave & 1) * 32;

  v8f acc[2][2];
#pragma unroll
  for (int i = 0; i < 2; ++i)
#pragma unroll
    for (int j = 0; j < 2; ++j)
#pragma unroll
      for (int e = 0; e < 8; ++e) acc[i][j][e] = 0.f;

  for (int k0 = 0; k0 < D_MODEL; k0 += 32) {
    v16h a0 = wmma_load_a(Xh + (size_t)m0 * D_MODEL + k0, D_MODEL, lane);
    v16h a1 = wmma_load_a(Xh + (size_t)(m0 + 16) * D_MODEL + k0, D_MODEL, lane);
    v16h b0 = wmma_load_bT(WaT + (size_t)n0 * D_MODEL + k0, D_MODEL, lane);
    v16h b1 = wmma_load_bT(WaT + (size_t)(n0 + 16) * D_MODEL + k0, D_MODEL, lane);
    acc[0][0] = wmma_f16(a0, b0, acc[0][0]);
    acc[0][1] = wmma_f16(a0, b1, acc[0][1]);
    acc[1][0] = wmma_f16(a1, b0, acc[1][0]);
    acc[1][1] = wmma_f16(a1, b1, acc[1][1]);
  }

  const int hs = lane >> 4;
  const int col = lane & 15;
#pragma unroll
  for (int ti = 0; ti < 2; ++ti)
#pragma unroll
    for (int tj = 0; tj < 2; ++tj)
#pragma unroll
      for (int v = 0; v < 8; ++v) {
        const int row = m0 + ti * 16 + v + hs * 8;
        const int n = n0 + tj * 16 + col;
        const float val = acc[ti][tj][v] + bias[n];
        const int which = n / D_MODEL;
        const int c = n - which * D_MODEL;
        const int h = c >> 6, d = c & 63;
        const int batch = row >> 12, t = row & (SEQ - 1);
        const size_t idx =
            (((size_t)batch * NUM_HEADS + h) * SEQ + t) * HEAD_DIM + d;
        if (which == 0)
          Qh[idx] = (_Float16)(val * 0.125f);  // fold 1/sqrt(64) into Q
        else if (which == 1)
          Kh[idx] = (_Float16)val;
        else
          Vh[idx] = (_Float16)val;
      }
}

// ---- async K/V chunk staging: 64 rows x 64 halves -> LDS (padded LDS_LD) --
__device__ __forceinline__ void async_copy_chunk(const _Float16* gK,
                                                 const _Float16* gV,
                                                 _Float16* lK, _Float16* lV,
                                                 int tid) {
  const int j = tid & 7;    // 16B chunk within a row
  const int r0 = tid >> 3;  // 0..15
#pragma unroll
  for (int i = 0; i < 4; ++i) {
    const int r = r0 + i * 16;
    const unsigned dK = (unsigned)(unsigned long long)(lK + r * LDS_LD + j * 8);
    const unsigned dV = (unsigned)(unsigned long long)(lV + r * LDS_LD + j * 8);
    const unsigned long long sK =
        (unsigned long long)(gK + (size_t)r * HEAD_DIM + j * 8);
    const unsigned long long sV =
        (unsigned long long)(gV + (size_t)r * HEAD_DIM + j * 8);
    asm volatile("global_load_async_to_lds_b128 %0, %1, off" ::"v"(dK), "v"(sK)
                 : "memory");
    asm volatile("global_load_async_to_lds_b128 %0, %1, off" ::"v"(dV), "v"(sV)
                 : "memory");
  }
}

// ---- Flash attention: 4 waves/block share LDS-staged K/V, online softmax --
__global__ __launch_bounds__(128) void flash_attn_kernel(
    const _Float16* __restrict__ Qh, const _Float16* __restrict__ Kh,
    const _Float16* __restrict__ Vh, _Float16* __restrict__ Oh) {
  __shared__ _Float16 lK[2][64 * LDS_LD];
  __shared__ _Float16 lV[2][64 * LDS_LD];
  __shared__ _Float16 lP[4][16 * LDS_LD];  // per-wave P staging (C->A layout)

  const int tid = threadIdx.x;
  const int lane = tid & 31;
  const int wave = tid >> 5;
  const int h = blockIdx.y, batch = blockIdx.z;
  const int q0 = blockIdx.x * 64 + wave * 16;

  const size_t bh = ((size_t)batch * NUM_HEADS + h) * SEQ * HEAD_DIM;
  const _Float16* Q = Qh + bh;
  const _Float16* K = Kh + bh;
  const _Float16* V = Vh + bh;

  v16h aq0 = wmma_load_a(Q + (size_t)q0 * HEAD_DIM + 0, HEAD_DIM, lane);
  v16h aq1 = wmma_load_a(Q + (size_t)q0 * HEAD_DIM + 32, HEAD_DIM, lane);

  v8f o[4];
  float mrow[8], lrow[8];
#pragma unroll
  for (int j = 0; j < 4; ++j)
#pragma unroll
    for (int e = 0; e < 8; ++e) o[j][e] = 0.f;
#pragma unroll
  for (int v = 0; v < 8; ++v) {
    mrow[v] = -3.0e38f;
    lrow[v] = 0.f;
  }

  const int hs = lane >> 4;
  const int col = lane & 15;
  _Float16* myp = &lP[wave][0];

  async_copy_chunk(K, V, lK[0], lV[0], tid);  // prime chunk 0

  for (int c = 0; c < SEQ / 64; ++c) {
    const int cur = c & 1;
    if (c + 1 < SEQ / 64) {
      // prefetch next chunk into the other buffer, then wait for the oldest 8
      // (current chunk) async ops: async loads complete in order.
      async_copy_chunk(K + (size_t)(c + 1) * 64 * HEAD_DIM,
                       V + (size_t)(c + 1) * 64 * HEAD_DIM, lK[cur ^ 1],
                       lV[cur ^ 1], tid);
      asm volatile("s_wait_asynccnt 8" ::: "memory");
    } else {
      asm volatile("s_wait_asynccnt 0" ::: "memory");
    }
    __syncthreads();  // chunk c staged by all waves
    const _Float16* Kc = lK[cur];
    const _Float16* Vc = lV[cur];

    // S = Q @ K^T for 16 queries x 64 keys (Q pre-scaled by 1/sqrt(D))
    v8f s[4];
#pragma unroll
    for (int j = 0; j < 4; ++j) {
      const _Float16* kp = Kc + 16 * j * LDS_LD;
      v16h bk0 = wmma_load_bT(kp + 0, LDS_LD, lane);
      v16h bk1 = wmma_load_bT(kp + 32, LDS_LD, lane);
      v8f z;
#pragma unroll
      for (int e = 0; e < 8; ++e) z[e] = 0.f;
      z = wmma_f16(aq0, bk0, z);
      s[j] = wmma_f16(aq1, bk1, z);
    }
    // online softmax: row max / rescale (reduce within 16-lane halves)
    float mnew[8], alpha[8];
#pragma unroll
    for (int v = 0; v < 8; ++v) {
      float m = s[0][v];
#pragma unroll
      for (int j = 1; j < 4; ++j) m = fmaxf(m, s[j][v]);
      m = fmaxf(m, __shfl_xor(m, 1, 32));
      m = fmaxf(m, __shfl_xor(m, 2, 32));
      m = fmaxf(m, __shfl_xor(m, 4, 32));
      m = fmaxf(m, __shfl_xor(m, 8, 32));
      mnew[v] = fmaxf(mrow[v], m);
      alpha[v] = __expf(mrow[v] - mnew[v]);
      mrow[v] = mnew[v];
    }
    // P = exp(S - mnew); row sums; stage P to LDS row-major for A operand
#pragma unroll
    for (int v = 0; v < 8; ++v) {
      float rs = 0.f;
#pragma unroll
      for (int j = 0; j < 4; ++j) {
        const float p = __expf(s[j][v] - mnew[v]);
        s[j][v] = p;
        rs += p;
      }
      rs += __shfl_xor(rs, 1, 32);
      rs += __shfl_xor(rs, 2, 32);
      rs += __shfl_xor(rs, 4, 32);
      rs += __shfl_xor(rs, 8, 32);
      lrow[v] = lrow[v] * alpha[v] + rs;
      const int prow = v + hs * 8;
#pragma unroll
      for (int j = 0; j < 4; ++j)
        myp[prow * LDS_LD + j * 16 + col] = (_Float16)s[j][v];
    }
    asm volatile("s_wait_dscnt 0" ::: "memory");  // LDS RAW (wave-private)
    v16h ap0 = wmma_load_a(myp + 0, LDS_LD, lane);
    v16h ap1 = wmma_load_a(myp + 32, LDS_LD, lane);
    // O = O*alpha + P @ V
#pragma unroll
    for (int jn = 0; jn < 4; ++jn) {
#pragma unroll
      for (int e = 0; e < 8; ++e) o[jn][e] *= alpha[e];
      v16h bv0 = wmma_load_b(Vc + 16 * jn, LDS_LD, lane);
      v16h bv1 = wmma_load_b(Vc + 32 * LDS_LD + 16 * jn, LDS_LD, lane);
      o[jn] = wmma_f16(ap0, bv0, o[jn]);
      o[jn] = wmma_f16(ap1, bv1, o[jn]);
    }
    __syncthreads();  // all waves done reading buf[cur] before it is re-filled
  }

  // normalize and write out to [B,T,C] f16 for the projection GEMM
  float inv[8];
#pragma unroll
  for (int v = 0; v < 8; ++v) inv[v] = 1.f / lrow[v];
#pragma unroll
  for (int jn = 0; jn < 4; ++jn)
#pragma unroll
    for (int v = 0; v < 8; ++v) {
      const int q = q0 + v + hs * 8;
      const int d = h * HEAD_DIM + jn * 16 + col;
      Oh[((size_t)batch * SEQ + q) * D_MODEL + d] = (_Float16)(o[jn][v] * inv[v]);
    }
}

// ---- Output projection: out = Oh @ W_proj + b_proj (f32 result) -----------
__global__ __launch_bounds__(128) void proj_gemm_kernel(
    const _Float16* __restrict__ Ah, const _Float16* __restrict__ WpT,
    const float* __restrict__ bias, float* __restrict__ out) {
  const int lane = threadIdx.x & 31;
  const int wave = threadIdx.x >> 5;
  const int m0 = blockIdx.y * 64 + (wave >> 1) * 32;
  const int n0 = blockIdx.x * 64 + (wave & 1) * 32;

  v8f acc[2][2];
#pragma unroll
  for (int i = 0; i < 2; ++i)
#pragma unroll
    for (int j = 0; j < 2; ++j)
#pragma unroll
      for (int e = 0; e < 8; ++e) acc[i][j][e] = 0.f;

  for (int k0 = 0; k0 < D_MODEL; k0 += 32) {
    v16h a0 = wmma_load_a(Ah + (size_t)m0 * D_MODEL + k0, D_MODEL, lane);
    v16h a1 = wmma_load_a(Ah + (size_t)(m0 + 16) * D_MODEL + k0, D_MODEL, lane);
    v16h b0 = wmma_load_bT(WpT + (size_t)n0 * D_MODEL + k0, D_MODEL, lane);
    v16h b1 = wmma_load_bT(WpT + (size_t)(n0 + 16) * D_MODEL + k0, D_MODEL, lane);
    acc[0][0] = wmma_f16(a0, b0, acc[0][0]);
    acc[0][1] = wmma_f16(a0, b1, acc[0][1]);
    acc[1][0] = wmma_f16(a1, b0, acc[1][0]);
    acc[1][1] = wmma_f16(a1, b1, acc[1][1]);
  }

  const int hs = lane >> 4;
  const int col = lane & 15;
#pragma unroll
  for (int ti = 0; ti < 2; ++ti)
#pragma unroll
    for (int tj = 0; tj < 2; ++tj)
#pragma unroll
      for (int v = 0; v < 8; ++v) {
        const int row = m0 + ti * 16 + v + hs * 8;
        const int n = n0 + tj * 16 + col;
        out[(size_t)row * D_MODEL + n] = acc[ti][tj][v] + bias[n];
      }
}

extern "C" void kernel_launch(void* const* d_in, const int* in_sizes, int n_in,
                              void* d_out, int out_size, void* d_ws,
                              size_t ws_size, hipStream_t stream) {
  const float* x = (const float*)d_in[0];
  const float* W_attn = (const float*)d_in[1];
  const float* b_attn = (const float*)d_in[2];
  const float* W_proj = (const float*)d_in[3];
  const float* b_proj = (const float*)d_in[4];
  float* out = (float*)d_out;

  char* ws = (char*)d_ws;
  size_t off = 0;
  auto alloc = [&](size_t bytes) {
    char* p = ws + off;
    off += (bytes + 255) & ~(size_t)255;
    return p;
  };
  _Float16* Xh = (_Float16*)alloc((size_t)M_ROWS * D_MODEL * 2);
  _Float16* WaT = (_Float16*)alloc((size_t)D_MODEL * N_QKV * 2);
  _Float16* WpT = (_Float16*)alloc((size_t)D_MODEL * D_MODEL * 2);
  _Float16* Qh = (_Float16*)alloc((size_t)M_ROWS * D_MODEL * 2);
  _Float16* Kh = (_Float16*)alloc((size_t)M_ROWS * D_MODEL * 2);
  _Float16* Vh = (_Float16*)alloc((size_t)M_ROWS * D_MODEL * 2);
  _Float16* Oh = (_Float16*)alloc((size_t)M_ROWS * D_MODEL * 2);
  if (off > ws_size) return;

  f32_to_f16_kernel<<<dim3(2048), 256, 0, stream>>>(x, Xh, M_ROWS * D_MODEL);
  transpose_f32_to_f16_kernel<<<dim3(1024), 256, 0, stream>>>(
      W_attn, WaT, D_MODEL, N_QKV);
  transpose_f32_to_f16_kernel<<<dim3(512), 256, 0, stream>>>(
      W_proj, WpT, D_MODEL, D_MODEL);
  qkv_gemm_kernel<<<dim3(N_QKV / 64, M_ROWS / 64), 128, 0, stream>>>(
      Xh, WaT, b_attn, Qh, Kh, Vh);
  flash_attn_kernel<<<dim3(SEQ / 64, NUM_HEADS, BATCH), 128, 0, stream>>>(
      Qh, Kh, Vh, Oh);
  proj_gemm_kernel<<<dim3(D_MODEL / 64, M_ROWS / 64), 128, 0, stream>>>(
      Oh, WpT, b_proj, out);
}
